// AdaptiveLowRank_14697378087041
// MI455X (gfx1250) — matrix operand order repaired and checked by reading, not compiled, not run on hardware
//
#include <hip/hip_runtime.h>
#include <hip/hip_bf16.h>
#include <math.h>

// ---------------------------------------------------------------------------
// AdaptiveLowRank on MI455X (gfx1250, wave32, WMMA)
//   probs = softmax(rank_weights[16]); W_eff = sum_r probs[r]*Ws[r]  (bf16)
//   out   = LayerNorm( (x*mask) @ W_eff^T )
// GEMM: M=8192, N=1024, K=1024 via v_wmma_f32_16x16x32_bf16.
// B operand is double-buffered through LDS with async global->LDS copies
// (ASYNCcnt pipeline); LayerNorm fused per 16-row block.
// ---------------------------------------------------------------------------

typedef __attribute__((ext_vector_type(16))) __bf16 v16bf;
typedef __attribute__((ext_vector_type(8)))  float  v8f;
typedef __attribute__((ext_vector_type(4)))  int    v4i;

#define B_DIM 4
#define S_DIM 2048
#define D_DIM 1024
#define R_NK  16
#define M_ROWS (B_DIM * S_DIM)      // 8192
#define OUT_N  (M_ROWS * D_DIM)     // 8388608 f32 elements, then 2 scalars

#if __has_builtin(__builtin_amdgcn_global_load_async_to_lds_b128) && \
    __has_builtin(__builtin_amdgcn_s_wait_asynccnt)
#define USE_ASYNC_LDS 1
#else
#define USE_ASYNC_LDS 0
#endif

__device__ __forceinline__ unsigned short f2bf_rne(float f) {
  unsigned int u = __float_as_uint(f);
  u += 0x7FFFu + ((u >> 16) & 1u);          // round-to-nearest-even to bf16
  return (unsigned short)(u >> 16);
}

#if USE_ASYNC_LDS
__device__ __forceinline__ void async_cp16(void* lds, const void* g) {
  __builtin_amdgcn_global_load_async_to_lds_b128(
      (__attribute__((address_space(1))) v4i*)g,
      (__attribute__((address_space(3))) v4i*)lds, 0, 0);
}
#endif

// --- kernel 1: softmax over 16 rank weights + scalar outputs ---------------
__global__ void k_setup(const float* __restrict__ rw,
                        float* __restrict__ probs,
                        float* __restrict__ out_scalars) {
  const int lane = threadIdx.x;               // 32 threads, one wave
  float w = (lane < R_NK) ? rw[lane] : -1e30f;
  float m = w;
  #pragma unroll
  for (int o = 16; o >= 1; o >>= 1) m = fmaxf(m, __shfl_xor(m, o, 32));
  float e = (lane < R_NK) ? expf(w - m) : 0.0f;
  float s = e;
  #pragma unroll
  for (int o = 16; o >= 1; o >>= 1) s += __shfl_xor(s, o, 32);
  float p = e / s;
  if (lane < R_NK) probs[lane] = p;
  float er  = p * (float)(lane + 1);                       // ranks 1..16
  float ent = (lane < R_NK) ? (-p * logf(p + 1e-8f)) : 0.0f;
  #pragma unroll
  for (int o = 16; o >= 1; o >>= 1) {
    er  += __shfl_xor(er,  o, 32);
    ent += __shfl_xor(ent, o, 32);
  }
  if (lane == 0) { out_scalars[0] = er; out_scalars[1] = ent; }
}

// --- kernel 2: W_eff = sum_r probs[r]*Ws[r], stored bf16 (float4 wide) -----
__global__ void __launch_bounds__(256) k_weff(const float* __restrict__ Ws,
                                              const float* __restrict__ probs,
                                              unsigned short* __restrict__ wb) {
  __shared__ float p[R_NK];
  if (threadIdx.x < R_NK) p[threadIdx.x] = probs[threadIdx.x];
  __syncthreads();
  const int i4 = blockIdx.x * blockDim.x + threadIdx.x;    // float4 index
  float4 acc = make_float4(0.f, 0.f, 0.f, 0.f);
  #pragma unroll
  for (int r = 0; r < R_NK; ++r) {
    const float4 v = ((const float4*)Ws)[(size_t)r * (D_DIM * D_DIM / 4) + i4];
    acc.x += p[r] * v.x; acc.y += p[r] * v.y;
    acc.z += p[r] * v.z; acc.w += p[r] * v.w;
  }
  ushort4 o;
  o.x = f2bf_rne(acc.x); o.y = f2bf_rne(acc.y);
  o.z = f2bf_rne(acc.z); o.w = f2bf_rne(acc.w);
  ((ushort4*)wb)[i4] = o;
}

// --- kernel 3: xb = bf16(x * mask) -----------------------------------------
__global__ void __launch_bounds__(256) k_xcvt(const float* __restrict__ x,
                                              const float* __restrict__ mask,
                                              unsigned short* __restrict__ xb) {
  const int i4 = blockIdx.x * blockDim.x + threadIdx.x;    // float4 index
  const float4 v = ((const float4*)x)[i4];
  const float  mk = mask[i4 >> 8];                         // (i4*4)/1024
  ushort4 o;
  o.x = f2bf_rne(v.x * mk);
  o.y = f2bf_rne(v.y * mk);
  o.z = f2bf_rne(v.z * mk);
  o.w = f2bf_rne(v.w * mk);
  ((ushort4*)xb)[i4] = o;
}

// --- B staging: wave-private 128x32 bf16 slice -> LDS (8 KB) ---------------
// 512 chunks of 16B; lane l copies chunks {i*32+l}. Row-major [128][32].
__device__ __forceinline__ void stage_B(const unsigned short* __restrict__ wb,
                                        int wave, int lane, int kb,
                                        uint4* __restrict__ buf) {
  const unsigned short* gsrc = wb + (size_t)(wave * 128) * D_DIM + kb;
  #pragma unroll
  for (int i = 0; i < 16; ++i) {
    const int c   = i * 32 + lane;
    const int row = c >> 2;
    const int sub = c & 3;
    const void* g = gsrc + (size_t)row * D_DIM + sub * 8;
    void*       l = (char*)buf + row * 64 + sub * 16;
#if USE_ASYNC_LDS
    async_cp16(l, g);
#else
    *(uint4*)l = *(const uint4*)g;
#endif
  }
}

// --- kernel 4: GEMM (16 rows x 1024 cols per block) + fused LayerNorm ------
// 8 waves; wave w owns columns [128w, 128w+128) as 8 WMMA tiles, shared A.
__global__ void __launch_bounds__(256) k_gemm_ln(const unsigned short* __restrict__ xb,
                                                 const unsigned short* __restrict__ wb,
                                                 const float* __restrict__ gamma,
                                                 const float* __restrict__ beta,
                                                 float* __restrict__ out) {
  // 128 KB: [0,64K) = B stage buf0 (aliased later by the f32 LN tile),
  //         [64K,128K) = B stage buf1. Wave w's region: 8 KB at w*512 uint4.
  __shared__ uint4 smem4[8192];

  const int tid   = threadIdx.x;
  const int wave  = tid >> 5;
  const int lane  = tid & 31;
  const int laneM = lane & 15;
  const int half  = lane >> 4;
  const int mBase = blockIdx.x * 16;

  uint4* stage0 = smem4 + wave * 512;          // 8 KB
  uint4* stage1 = smem4 + 4096 + wave * 512;   // 8 KB

  const unsigned short* arow = xb + (size_t)(mBase + laneM) * D_DIM;

  v8f acc[8];
  #pragma unroll
  for (int t = 0; t < 8; ++t) acc[t] = (v8f){0.f,0.f,0.f,0.f,0.f,0.f,0.f,0.f};

  union Frag { uint4 u[2]; v16bf v; };

  stage_B(wb, wave, lane, 0, stage0);          // prime the async pipeline

  for (int s = 0; s < 32; ++s) {
    const int kb = s * 32;
    uint4* cur = (s & 1) ? stage1 : stage0;
    uint4* nxt = (s & 1) ? stage0 : stage1;

    if (s + 1 < 32) {
      stage_B(wb, wave, lane, kb + 32, nxt);   // 16 async copies in flight
#if USE_ASYNC_LDS
      __builtin_amdgcn_s_wait_asynccnt(16);    // cur buffer complete
#endif
      __builtin_prefetch(arow + kb + 32, 0, 1); // global_prefetch_b8 next A
    } else {
#if USE_ASYNC_LDS
      __builtin_amdgcn_s_wait_asynccnt(0);
#endif
    }

    // A fragment (16x32 bf16): elems 0-7 <- K kb+8h.., elems 8-15 <- kb+16+8h
    Frag a;
    a.u[0] = *(const uint4*)(arow + kb + 8 * half);
    a.u[1] = *(const uint4*)(arow + kb + 16 + 8 * half);

    const char* bs = (const char*)cur;
    #pragma unroll
    for (int t = 0; t < 8; ++t) {
      // B fragment (32x16 bf16): elems 0-7 <- K 8h.., elems 8-15 <- 16+8h, row n=t*16+laneM
      Frag b;
      const int boff = (t * 16 + laneM) * 64 + 16 * half;
      b.u[0] = *(const uint4*)(bs + boff);
      b.u[1] = *(const uint4*)(bs + boff + 32);
      acc[t] = __builtin_amdgcn_wmma_f32_16x16x32_bf16(
          false, a.v, false, b.v, (short)0, acc[t], false, false);
    }
  }

  // staging buffers now dead everywhere -> reuse [0,64K) as the f32 LN tile
  __syncthreads();
  float (*tile)[D_DIM] = (float(*)[D_DIM])smem4;

  // scatter C (VGPR r, lane) = D[r + 8*half][laneM] into the LDS row tile
  #pragma unroll
  for (int t = 0; t < 8; ++t) {
    const int col = wave * 128 + t * 16 + laneM;
    #pragma unroll
    for (int r = 0; r < 8; ++r)
      tile[8 * half + r][col] = acc[t][r];
  }
  __syncthreads();

  // LayerNorm: each wave normalizes rows 2*wave and 2*wave+1
  #pragma unroll
  for (int rr = 0; rr < 2; ++rr) {
    const int row = wave * 2 + rr;
    float s = 0.f, sq = 0.f;
    #pragma unroll
    for (int i = 0; i < D_DIM / 32; ++i) {
      const float v = tile[row][lane + 32 * i];
      s += v; sq += v * v;
    }
    #pragma unroll
    for (int o = 16; o >= 1; o >>= 1) {
      s  += __shfl_xor(s,  o, 32);
      sq += __shfl_xor(sq, o, 32);
    }
    const float mean = s * (1.0f / D_DIM);
    const float var  = sq * (1.0f / D_DIM) - mean * mean;
    const float rstd = rsqrtf(var + 1e-5f);
    float* orow = out + (size_t)(mBase + row) * D_DIM;
    #pragma unroll 4
    for (int i = 0; i < D_DIM / 32; ++i) {
      const int col = lane + 32 * i;
      orow[col] = (tile[row][col] - mean) * rstd * gamma[col] + beta[col];
    }
  }
}

// ---------------------------------------------------------------------------
extern "C" void kernel_launch(void* const* d_in, const int* in_sizes, int n_in,
                              void* d_out, int out_size, void* d_ws, size_t ws_size,
                              hipStream_t stream) {
  (void)in_sizes; (void)n_in; (void)out_size; (void)ws_size;

  const float* x     = (const float*)d_in[0];   // [4,2048,1024]
  const float* mask  = (const float*)d_in[1];   // [4,2048]
  const float* rw    = (const float*)d_in[2];   // [16]
  const float* Ws    = (const float*)d_in[3];   // [16,1024,1024]
  const float* gamma = (const float*)d_in[4];   // [1024]
  const float* beta  = (const float*)d_in[5];   // [1024]
  float*       out   = (float*)d_out;           // [8388608] + 2 scalars

  // workspace layout
  float*          probs = (float*)d_ws;                                   // 64 B
  unsigned short* xb    = (unsigned short*)((char*)d_ws + 1024);          // 16 MB
  unsigned short* wb    = (unsigned short*)((char*)d_ws + 1024 + (size_t)16*1024*1024); // 2 MB

  k_setup<<<1, 32, 0, stream>>>(rw, probs, out + OUT_N);
  k_weff<<<(D_DIM * D_DIM / 4) / 256, 256, 0, stream>>>(Ws, probs, wb);
  k_xcvt<<<(M_ROWS * D_DIM / 4) / 256, 256, 0, stream>>>(x, mask, xb);
  k_gemm_ln<<<M_ROWS / 16, 256, 0, stream>>>(xb, wb, gamma, beta, out);
}
